// GeometricCrossAttentionBlock_56831007261072
// MI455X (gfx1250) — matrix operand order repaired and checked by reading, not compiled, not run on hardware
//
#include <hip/hip_runtime.h>
#include <hip/hip_bf16.h>
#include <math.h>

typedef __attribute__((ext_vector_type(16))) _Float16 v16h;
typedef __attribute__((ext_vector_type(8)))  float    v8f;

union FragA {
    v16h v;
    _Float16 h[16];
    uint4 u[2];
};

__device__ inline v8f wmma16x16x32(const FragA& a, const FragA& b, v8f c) {
    // D = A(16x32 f16) * B(32x16 f16) + C(16x16 f32)
    return __builtin_amdgcn_wmma_f32_16x16x32_f16(
        /*neg_a=*/false, a.v, /*neg_b=*/false, b.v,
        /*c_mod=*/(short)0, c, /*reuse_a=*/false, /*reuse_b=*/false);
}

__device__ inline float gelu_exact(float x) {
    return 0.5f * x * (1.0f + erff(x * 0.70710678118654752440f));
}

// CDNA5 async global->LDS copies (ASYNCcnt-tracked, no VGPR round trip).
// Low 32 bits of a generic pointer to __shared__ are the LDS byte offset
// (ISA 10.2: LDS aperture carries the offset in addr[31:0]).
__device__ inline void async_ld_b128(const _Float16* g, _Float16* l) {
    unsigned loff = (unsigned)(uintptr_t)l;
    unsigned long long ga = (unsigned long long)(uintptr_t)g;
    asm volatile("global_load_async_to_lds_b128 %0, %1, off"
                 :: "v"(loff), "v"(ga) : "memory");
}
__device__ inline void async_ld_b64(const _Float16* g, _Float16* l) {
    unsigned loff = (unsigned)(uintptr_t)l;
    unsigned long long ga = (unsigned long long)(uintptr_t)g;
    asm volatile("global_load_async_to_lds_b64 %0, %1, off"
                 :: "v"(loff), "v"(ga) : "memory");
}
__device__ inline void wait_async0() {
    asm volatile("s_wait_asynccnt 0x0" ::: "memory");
}

// ---------------------------------------------------------------------------
// LayerNorm over last dim (=256), fp32 in -> f16 out. One block per row.
// ---------------------------------------------------------------------------
__global__ void ln_kernel(const float* __restrict__ x, const float* __restrict__ gam,
                          const float* __restrict__ bet, _Float16* __restrict__ y) {
    __shared__ float s1[256];
    __shared__ float s2[256];
    const int row = blockIdx.x;
    const int t = threadIdx.x;
    float v = x[(size_t)row * 256 + t];
    s1[t] = v;
    s2[t] = v * v;
    __syncthreads();
    for (int s = 128; s > 0; s >>= 1) {
        if (t < s) { s1[t] += s1[t + s]; s2[t] += s2[t + s]; }
        __syncthreads();
    }
    const float mu  = s1[0] * (1.0f / 256.0f);
    const float var = s2[0] * (1.0f / 256.0f) - mu * mu;
    const float inv = rsqrtf(var + 1e-5f);
    y[(size_t)row * 256 + t] = (_Float16)((v - mu) * inv * gam[t] + bet[t]);
}

// ---------------------------------------------------------------------------
// GEMM: out[M,Ncols] = act( A[M,K](f16) @ W[K,Ncols](f32) + bias (+ res) )
// 64x64 tile/block, 8 waves, wave owns 16x32 via two WMMA accumulators.
// A tile staged via async global->LDS B128 (1 per thread); W^T staged via
// VALU (fp32 load, cvt, transposed ds stores).
// ---------------------------------------------------------------------------
template<bool GELU, bool RES, bool OUTF16>
__global__ void gemm_kernel(const _Float16* __restrict__ Ag, const float* __restrict__ Wg,
                            const float* __restrict__ bias, const float* __restrict__ res,
                            void* __restrict__ outp, int K, int Ncols) {
    constexpr int RS = 40;  // padded row stride in halves (80B, keeps 16B alignment)
    __shared__ alignas(16) _Float16 lA[64 * RS];   // [m][k] f16
    __shared__ alignas(16) _Float16 lB[64 * RS];   // [n][k] f16  (= W^T tile)

    const int tid  = threadIdx.x;
    const int lane = tid & 31;
    const int w    = tid >> 5;
    const int wm   = w >> 1;          // 0..3
    const int wn   = w & 1;           // 0..1
    const int m    = lane & 15;
    const int g    = lane >> 4;

    const int row0 = blockIdx.x * 64;
    const int col0 = blockIdx.y * 64;

    // Per-thread async A-copy coords: 64 rows x 64B; 256 threads x 16B.
    const int arow = tid >> 2;        // 0..63
    const int achk = tid & 3;         // 16B chunk within row

    v8f acc0 = {0.f,0.f,0.f,0.f,0.f,0.f,0.f,0.f};
    v8f acc1 = {0.f,0.f,0.f,0.f,0.f,0.f,0.f,0.f};

    for (int k0 = 0; k0 < K; k0 += 32) {
        // A tile 64x32 f16: one async B128 per thread.
        async_ld_b128(Ag + (size_t)(row0 + arow) * K + k0 + achk * 8,
                      &lA[arow * RS + achk * 8]);
        // W^T tile 64x32: fp32 load + cvt + transposed store.
        #pragma unroll
        for (int e = 0; e < 8; ++e) {
            int fl = tid + e * 256;
            int bn = fl & 63, bk = fl >> 6;
            lB[bn * RS + bk] = (_Float16)Wg[(size_t)(k0 + bk) * Ncols + col0 + bn];
        }
        wait_async0();
        __syncthreads();

        FragA fa, fb0, fb1;
        const _Float16* pa = &lA[(wm * 16 + m) * RS];
        fa.u[0] = *(const uint4*)(pa + g * 16);
        fa.u[1] = *(const uint4*)(pa + g * 16 + 8);
        const _Float16* pb0 = &lB[(wn * 32 + m) * RS + g * 16];
        fb0.u[0] = *(const uint4*)(pb0);
        fb0.u[1] = *(const uint4*)(pb0 + 8);
        const _Float16* pb1 = &lB[(wn * 32 + 16 + m) * RS + g * 16];
        fb1.u[0] = *(const uint4*)(pb1);
        fb1.u[1] = *(const uint4*)(pb1 + 8);

        acc0 = wmma16x16x32(fa, fb0, acc0);
        acc1 = wmma16x16x32(fa, fb1, acc1);
        __syncthreads();
    }

    #pragma unroll
    for (int r = 0; r < 8; ++r) {
        const int row = row0 + wm * 16 + r + 8 * g;
        const int c0  = col0 + wn * 32 + m;
        const int c1  = c0 + 16;
        float v0 = acc0[r] + bias[c0];
        float v1 = acc1[r] + bias[c1];
        if (GELU) { v0 = gelu_exact(v0); v1 = gelu_exact(v1); }
        if (RES)  { v0 += res[(size_t)row * Ncols + c0]; v1 += res[(size_t)row * Ncols + c1]; }
        if (OUTF16) {
            _Float16* out = (_Float16*)outp;
            out[(size_t)row * Ncols + c0] = (_Float16)v0;
            out[(size_t)row * Ncols + c1] = (_Float16)v1;
        } else {
            float* out = (float*)outp;
            out[(size_t)row * Ncols + c0] = v0;
            out[(size_t)row * Ncols + c1] = v1;
        }
    }
}

// ---------------------------------------------------------------------------
// Fused flash-style attention for one (b,h) per blockIdx.y; q,k,v f16
// [B*N,256] with head h in columns h*32..+31. Block = 128 q-rows (8 waves).
// K tile staged via async B64 copies; V transposed via VALU/DS; online
// softmax with xor-shuffle row reductions; PV via WMMA; f16 output.
// ---------------------------------------------------------------------------
template<bool HASBIAS>
__global__ void attn_kernel(const _Float16* __restrict__ q, const _Float16* __restrict__ k,
                            const _Float16* __restrict__ v, const float* __restrict__ bias,
                            _Float16* __restrict__ out, int N, int H, float scale) {
    constexpr int RS = 40;
    __shared__ alignas(16) _Float16 lK[32 * RS];    // [key][dim]  (= B^T for Q@K^T)
    __shared__ alignas(16) _Float16 lVT[32 * RS];   // [dim][key]  (= B^T for P@V)
    __shared__ alignas(16) _Float16 lP[8 * 16 * 32];// per-wave P tile, [qrow][key]

    const int tid  = threadIdx.x;
    const int lane = tid & 31;
    const int w    = tid >> 5;
    const int m    = lane & 15;
    const int g    = lane >> 4;

    const int bh = blockIdx.y;
    const int b  = bh / H;
    const int h  = bh % H;
    const int qi0 = blockIdx.x * 128 + w * 16;
    const size_t qrow0 = (size_t)b * N + qi0;
    const int co = h * 32;

    // Staging coords: 32 keys x 64B rows; 256 threads x 8B for K (async B64),
    // 256 threads x 4 halves for the V transpose.
    const int skey = tid >> 3;        // 0..31
    const int schk = tid & 7;         // 8B chunk within row

    // Q fragment straight from global (A-layout runs are 16B-contiguous).
    FragA fq;
    {
        const _Float16* pq = q + (qrow0 + m) * 256 + co;
        fq.u[0] = *(const uint4*)(pq + 16 * g);
        fq.u[1] = *(const uint4*)(pq + 16 * g + 8);
    }

    v8f acc0 = {0.f,0.f,0.f,0.f,0.f,0.f,0.f,0.f};
    v8f acc1 = {0.f,0.f,0.f,0.f,0.f,0.f,0.f,0.f};
    float rm[8], rl[8];
    #pragma unroll
    for (int r = 0; r < 8; ++r) { rm[r] = -1e30f; rl[r] = 0.0f; }

    _Float16* myP = &lP[w * 512];

    for (int kb = 0; kb < N; kb += 32) {
        const size_t grow = (size_t)b * N + kb + skey;
        // K tile: async copy, no VGPR round trip.
        async_ld_b64(k + grow * 256 + co + schk * 4, &lK[skey * RS + schk * 4]);
        // V tile transposed: 4 halves per thread.
        {
            union { unsigned long long u; _Float16 hh[4]; } vv;
            vv.u = *(const unsigned long long*)(v + grow * 256 + co + schk * 4);
            #pragma unroll
            for (int i = 0; i < 4; ++i)
                lVT[(schk * 4 + i) * RS + skey] = vv.hh[i];
        }
        wait_async0();
        __syncthreads();

        // S = Q @ K^T for two 16-key sub-tiles.
        FragA fk0, fk1;
        const _Float16* pk0 = &lK[m * RS + g * 16];
        fk0.u[0] = *(const uint4*)pk0;  fk0.u[1] = *(const uint4*)(pk0 + 8);
        const _Float16* pk1 = &lK[(16 + m) * RS + g * 16];
        fk1.u[0] = *(const uint4*)pk1;  fk1.u[1] = *(const uint4*)(pk1 + 8);

        v8f zero = {0.f,0.f,0.f,0.f,0.f,0.f,0.f,0.f};
        v8f s0 = wmma16x16x32(fq, fk0, zero);
        v8f s1 = wmma16x16x32(fq, fk1, zero);

        #pragma unroll
        for (int r = 0; r < 8; ++r) {
            if (HASBIAS) {
                const int qi = qi0 + r + 8 * g;
                const size_t base = ((size_t)bh * N + qi) * (size_t)N + kb + m;
                s0[r] = fmaf(s0[r], scale, bias[base]);
                s1[r] = fmaf(s1[r], scale, bias[base + 16]);
            } else {
                s0[r] *= scale;
                s1[r] *= scale;
            }
        }

        // Online softmax; D-layout row = r + 8g, 16 lanes per row group.
        #pragma unroll
        for (int r = 0; r < 8; ++r) {
            float t = fmaxf(s0[r], s1[r]);
            t = fmaxf(t, __shfl_xor(t, 1, 32));
            t = fmaxf(t, __shfl_xor(t, 2, 32));
            t = fmaxf(t, __shfl_xor(t, 4, 32));
            t = fmaxf(t, __shfl_xor(t, 8, 32));
            const float mn    = fmaxf(rm[r], t);
            const float alpha = __expf(rm[r] - mn);
            rm[r] = mn;
            const float p0 = __expf(s0[r] - mn);
            const float p1 = __expf(s1[r] - mn);
            float ps = p0 + p1;
            ps += __shfl_xor(ps, 1, 32);
            ps += __shfl_xor(ps, 2, 32);
            ps += __shfl_xor(ps, 4, 32);
            ps += __shfl_xor(ps, 8, 32);
            rl[r] = rl[r] * alpha + ps;
            acc0[r] *= alpha;
            acc1[r] *= alpha;
            myP[(r + 8 * g) * 32 + m]      = (_Float16)p0;
            myP[(r + 8 * g) * 32 + 16 + m] = (_Float16)p1;
        }
        // Wave-local LDS visibility before cross-lane readback.
        asm volatile("s_wait_dscnt 0x0" ::: "memory");

        FragA fp, fv0, fv1;
        const _Float16* pp = &myP[m * 32];
        fp.u[0] = *(const uint4*)(pp + g * 16);
        fp.u[1] = *(const uint4*)(pp + g * 16 + 8);
        const _Float16* pv0 = &lVT[m * RS + g * 16];
        fv0.u[0] = *(const uint4*)pv0;  fv0.u[1] = *(const uint4*)(pv0 + 8);
        const _Float16* pv1 = &lVT[(16 + m) * RS + g * 16];
        fv1.u[0] = *(const uint4*)pv1;  fv1.u[1] = *(const uint4*)(pv1 + 8);

        acc0 = wmma16x16x32(fp, fv0, acc0);   // dims 0..15
        acc1 = wmma16x16x32(fp, fv1, acc1);   // dims 16..31
        __syncthreads();
    }

    #pragma unroll
    for (int r = 0; r < 8; ++r) {
        const size_t row = qrow0 + r + 8 * g;
        const float invl = 1.0f / rl[r];
        out[row * 256 + co + m]      = (_Float16)(acc0[r] * invl);
        out[row * 256 + co + 16 + m] = (_Float16)(acc1[r] * invl);
    }
}

// ---------------------------------------------------------------------------
extern "C" void kernel_launch(void* const* d_in, const int* in_sizes, int n_in,
                              void* d_out, int out_size, void* d_ws, size_t ws_size,
                              hipStream_t stream) {
    (void)in_sizes; (void)n_in; (void)out_size; (void)ws_size;

    const float* src_feats = (const float*)d_in[0];
    const float* tgt_feats = (const float*)d_in[1];
    const float* src_gb    = (const float*)d_in[2];
    const float* tgt_gb    = (const float*)d_in[3];
    const float* sa_q_w = (const float*)d_in[4];  const float* sa_q_b = (const float*)d_in[5];
    const float* sa_k_w = (const float*)d_in[6];  const float* sa_k_b = (const float*)d_in[7];
    const float* sa_v_w = (const float*)d_in[8];  const float* sa_v_b = (const float*)d_in[9];
    const float* sa_o_w = (const float*)d_in[10]; const float* sa_o_b = (const float*)d_in[11];
    const float* ca_q_w = (const float*)d_in[12]; const float* ca_q_b = (const float*)d_in[13];
    const float* ca_k_w = (const float*)d_in[14]; const float* ca_k_b = (const float*)d_in[15];
    const float* ca_v_w = (const float*)d_in[16]; const float* ca_v_b = (const float*)d_in[17];
    const float* ca_o_w = (const float*)d_in[18]; const float* ca_o_b = (const float*)d_in[19];
    const float* ln_sa_g = (const float*)d_in[20]; const float* ln_sa_b = (const float*)d_in[21];
    const float* ln_ca_g = (const float*)d_in[22]; const float* ln_ca_b = (const float*)d_in[23];
    const float* ln_ff_g = (const float*)d_in[24]; const float* ln_ff_b = (const float*)d_in[25];
    const float* ffn_w1 = (const float*)d_in[26]; const float* ffn_b1 = (const float*)d_in[27];
    const float* ffn_w2 = (const float*)d_in[28]; const float* ffn_b2 = (const float*)d_in[29];

    constexpr int Bb = 4, Nn = 1024, Dd = 256, Hh = 8, Ff = 1024;
    constexpr int R  = Bb * Nn;
    constexpr size_t A  = (size_t)R * Dd;      // elements per [R,256] buffer
    constexpr size_t MB = 1u << 20;
    const float scale = 0.17677669529663687f;  // 32^-0.5

    char* wsb = (char*)d_ws;
    float*    cur_s = (float*)   (wsb + 0 * MB);   // fp32, 4 MiB
    float*    cur_t = (float*)   (wsb + 4 * MB);
    _Float16* xn_s  = (_Float16*)(wsb + 8 * MB);   // f16, 2 MiB each
    _Float16* xn_t  = (_Float16*)(wsb + 10 * MB);
    _Float16* q_s   = (_Float16*)(wsb + 12 * MB);
    _Float16* k_s   = (_Float16*)(wsb + 14 * MB);
    _Float16* v_s   = (_Float16*)(wsb + 16 * MB);
    _Float16* q_t   = (_Float16*)(wsb + 18 * MB);
    _Float16* k_t   = (_Float16*)(wsb + 20 * MB);
    _Float16* v_t   = (_Float16*)(wsb + 22 * MB);
    _Float16* ao_s  = (_Float16*)(wsb + 24 * MB);
    _Float16* ao_t  = (_Float16*)(wsb + 26 * MB);
    _Float16* ffh   = (_Float16*)(wsb + 12 * MB);  // [4096,1024] f16, reuses q/k/v region
    float* out_s = (float*)d_out;
    float* out_t = out_s + A;

    dim3 blk(256);
    dim3 gLN(R);
    dim3 gG (R / 64, Dd / 64);
    dim3 gG1(R / 64, Ff / 64);
    dim3 gAtt(Nn / 128, Bb * Hh);

    // ---------------- Stage 1: geometric self-attention ----------------
    ln_kernel<<<gLN, blk, 0, stream>>>(src_feats, ln_sa_g, ln_sa_b, xn_s);
    ln_kernel<<<gLN, blk, 0, stream>>>(tgt_feats, ln_sa_g, ln_sa_b, xn_t);

    gemm_kernel<false,false,true><<<gG, blk, 0, stream>>>(xn_s, sa_q_w, sa_q_b, nullptr, q_s, Dd, Dd);
    gemm_kernel<false,false,true><<<gG, blk, 0, stream>>>(xn_s, sa_k_w, sa_k_b, nullptr, k_s, Dd, Dd);
    gemm_kernel<false,false,true><<<gG, blk, 0, stream>>>(xn_s, sa_v_w, sa_v_b, nullptr, v_s, Dd, Dd);
    gemm_kernel<false,false,true><<<gG, blk, 0, stream>>>(xn_t, sa_q_w, sa_q_b, nullptr, q_t, Dd, Dd);
    gemm_kernel<false,false,true><<<gG, blk, 0, stream>>>(xn_t, sa_k_w, sa_k_b, nullptr, k_t, Dd, Dd);
    gemm_kernel<false,false,true><<<gG, blk, 0, stream>>>(xn_t, sa_v_w, sa_v_b, nullptr, v_t, Dd, Dd);

    attn_kernel<true><<<gAtt, blk, 0, stream>>>(q_s, k_s, v_s, src_gb, ao_s, Nn, Hh, scale);
    attn_kernel<true><<<gAtt, blk, 0, stream>>>(q_t, k_t, v_t, tgt_gb, ao_t, Nn, Hh, scale);

    gemm_kernel<false,true,false><<<gG, blk, 0, stream>>>(ao_s, sa_o_w, sa_o_b, src_feats, cur_s, Dd, Dd);
    gemm_kernel<false,true,false><<<gG, blk, 0, stream>>>(ao_t, sa_o_w, sa_o_b, tgt_feats, cur_t, Dd, Dd);

    // ---------------- Stage 2: cross-attention ----------------
    ln_kernel<<<gLN, blk, 0, stream>>>(cur_s, ln_ca_g, ln_ca_b, xn_s);
    ln_kernel<<<gLN, blk, 0, stream>>>(cur_t, ln_ca_g, ln_ca_b, xn_t);

    gemm_kernel<false,false,true><<<gG, blk, 0, stream>>>(xn_s, ca_q_w, ca_q_b, nullptr, q_s, Dd, Dd);
    gemm_kernel<false,false,true><<<gG, blk, 0, stream>>>(xn_t, ca_k_w, ca_k_b, nullptr, k_t, Dd, Dd);
    gemm_kernel<false,false,true><<<gG, blk, 0, stream>>>(xn_t, ca_v_w, ca_v_b, nullptr, v_t, Dd, Dd);
    gemm_kernel<false,false,true><<<gG, blk, 0, stream>>>(xn_t, ca_q_w, ca_q_b, nullptr, q_t, Dd, Dd);
    gemm_kernel<false,false,true><<<gG, blk, 0, stream>>>(xn_s, ca_k_w, ca_k_b, nullptr, k_s, Dd, Dd);
    gemm_kernel<false,false,true><<<gG, blk, 0, stream>>>(xn_s, ca_v_w, ca_v_b, nullptr, v_s, Dd, Dd);

    attn_kernel<false><<<gAtt, blk, 0, stream>>>(q_s, k_t, v_t, nullptr, ao_s, Nn, Hh, scale);
    attn_kernel<false><<<gAtt, blk, 0, stream>>>(q_t, k_s, v_s, nullptr, ao_t, Nn, Hh, scale);

    gemm_kernel<false,true,false><<<gG, blk, 0, stream>>>(ao_s, ca_o_w, ca_o_b, cur_s, cur_s, Dd, Dd);
    gemm_kernel<false,true,false><<<gG, blk, 0, stream>>>(ao_t, ca_o_w, ca_o_b, cur_t, cur_t, Dd, Dd);

    // ---------------- Stage 3: FFN (src then tgt; ffh reused) ----------------
    ln_kernel<<<gLN, blk, 0, stream>>>(cur_s, ln_ff_g, ln_ff_b, xn_s);
    gemm_kernel<true,false,true ><<<gG1, blk, 0, stream>>>(xn_s, ffn_w1, ffn_b1, nullptr, ffh, Dd, Ff);
    gemm_kernel<false,true,false><<<gG,  blk, 0, stream>>>(ffh, ffn_w2, ffn_b2, cur_s, out_s, Ff, Dd);

    ln_kernel<<<gLN, blk, 0, stream>>>(cur_t, ln_ff_g, ln_ff_b, xn_t);
    gemm_kernel<true,false,true ><<<gG1, blk, 0, stream>>>(xn_t, ffn_w1, ffn_b1, nullptr, ffh, Dd, Ff);
    gemm_kernel<false,true,false><<<gG,  blk, 0, stream>>>(ffh, ffn_w2, ffn_b2, cur_t, out_t, Ff, Dd);
}
